// GPTBlock_46377056862599
// MI455X (gfx1250) — compile-verified, matrix operand order checked by reading, not run on hardware
//
#include <hip/hip_runtime.h>
#include <math.h>

// ---------------- types ----------------
typedef unsigned short u16;
typedef __attribute__((ext_vector_type(16))) __bf16 v16bf;
typedef __attribute__((ext_vector_type(8)))  float  v8f;
typedef __attribute__((ext_vector_type(4)))  int    v4i;

#define E_    1024
#define T3_   3072
#define HEADS 16
#define HD    64
#define SEQ   2048
#define BATCH 4
#define NTOK  (BATCH*SEQ)   // 8192

#if __has_builtin(__builtin_amdgcn_global_load_async_to_lds_b128)
#define HAVE_ASYNC_LDS 1
#endif

// f32 -> bf16 (round to nearest even), raw bits
__device__ inline u16 f2bf(float f) {
  unsigned u = __builtin_bit_cast(unsigned, f);
  unsigned r = u + 0x7fffu + ((u >> 16) & 1u);
  return (u16)(r >> 16);
}
__device__ inline unsigned pack2bf(float a, float b) {
  return (unsigned)f2bf(a) | ((unsigned)f2bf(b) << 16);
}

struct Frag16B { uint4 lo; uint4 hi; };

// Build a v16bf WMMA fragment from two contiguous 8x bf16 (16B) chunks.
// CDNA5 16-bit A 16x32 / B 32x16 layout: lane holds row/col = lane%16,
// K chunks at g*8 and 16+g*8, g = lane/16.
__device__ inline v16bf load_frag(const u16* p0, const u16* p1) {
  Frag16B f;
  f.lo = *(const uint4*)p0;
  f.hi = *(const uint4*)p1;
  return __builtin_bit_cast(v16bf, f);
}

__device__ inline v8f v8f_zero() {
  v8f z = {0.f,0.f,0.f,0.f,0.f,0.f,0.f,0.f};
  return z;
}

// 16B global -> LDS copy; async DMA (ASYNCcnt) when the gfx1250 builtin exists.
__device__ inline void copy16_g2l(const u16* g, u16* l) {
#ifdef HAVE_ASYNC_LDS
  __builtin_amdgcn_global_load_async_to_lds_b128((v4i*)g, (v4i*)l, 0, 0);
#else
  *(uint4*)l = *(const uint4*)g;
#endif
}
__device__ inline void async_copies_wait() {
#ifdef HAVE_ASYNC_LDS
#if __has_builtin(__builtin_amdgcn_s_wait_asynccnt)
  __builtin_amdgcn_s_wait_asynccnt(0);
#else
  asm volatile("s_wait_asynccnt 0" ::: "memory");
#endif
#endif
}

// ---------------- LayerNorm -> bf16 ----------------
__global__ __launch_bounds__(256)
void layernorm_bf16_kernel(const float* __restrict__ x, const float* __restrict__ g,
                           const float* __restrict__ b, u16* __restrict__ out)
{
  __shared__ float red[256];
  const int row = blockIdx.x;
  const int tid = threadIdx.x;
  const float* xr = x + (size_t)row * E_;
  float vals[4];
  float s = 0.f;
  #pragma unroll
  for (int i = 0; i < 4; i++) { vals[i] = xr[tid + 256*i]; s += vals[i]; }
  red[tid] = s; __syncthreads();
  for (int off = 128; off > 0; off >>= 1) {
    if (tid < off) red[tid] += red[tid + off];
    __syncthreads();
  }
  const float mu = red[0] * (1.f / E_);
  __syncthreads();
  s = 0.f;
  #pragma unroll
  for (int i = 0; i < 4; i++) { float d = vals[i] - mu; s += d * d; }
  red[tid] = s; __syncthreads();
  for (int off = 128; off > 0; off >>= 1) {
    if (tid < off) red[tid] += red[tid + off];
    __syncthreads();
  }
  const float rstd = rsqrtf(red[0] * (1.f / E_) + 1e-5f);
  #pragma unroll
  for (int i = 0; i < 4; i++) {
    int c = tid + 256*i;
    float v = (vals[i] - mu) * rstd * g[c] + b[c];
    out[(size_t)row * E_ + c] = f2bf(v);
  }
}

// ---------------- bf16 WMMA GEMM: C = A(bf16) @ B(f32->bf16) + epilogue ----------------
enum { EPI_QKV = 0, EPI_RES = 1, EPI_GELU = 2 };

template<int EPI>
__global__ __launch_bounds__(128)
void gemm_bf16_kernel(const u16* __restrict__ A, const float* __restrict__ Bw,
                      const float* __restrict__ bias, const float* __restrict__ res,
                      u16* __restrict__ outB, float* __restrict__ outF,
                      int M, int N, int K)
{
  __shared__ u16 ldsA [64 * 64];   // [row][k]   8 KB
  __shared__ u16 ldsBt[64 * 64];   // [n][k]     8 KB (B transposed)
  const int tid  = threadIdx.x;
  const int wave = tid >> 5, lane = tid & 31;
  const int lr   = lane & 15, lhi = lane >> 4;
  const int m0 = blockIdx.y * 64, n0 = blockIdx.x * 64;

  v8f acc[4];
  #pragma unroll
  for (int i = 0; i < 4; i++) acc[i] = v8f_zero();

  for (int k0 = 0; k0 < K; k0 += 64) {
    __syncthreads();
    // A tile 64x64 bf16: 4x 16B async copies per thread
    #pragma unroll
    for (int q = 0; q < 4; q++) {
      int c = tid * 4 + q;                 // 512 chunks
      int r = c >> 3, off = (c & 7) * 8;   // 8 chunks per 64-elem row
      copy16_g2l(&A[(size_t)(m0 + r) * K + k0 + off], &ldsA[r * 64 + off]);
    }
    // B tile 64x64 f32 -> bf16, transposed into [n][k], pair-packed b32 stores
    #pragma unroll
    for (int i = 0; i < 16; i++) {
      int lin = tid + 128 * i;             // 0..2047
      int nn = lin & 63, kk = (lin >> 6) * 2;
      float v0 = Bw[(size_t)(k0 + kk)     * N + n0 + nn];
      float v1 = Bw[(size_t)(k0 + kk + 1) * N + n0 + nn];
      *(unsigned*)&ldsBt[nn * 64 + kk] = pack2bf(v0, v1);
    }
    async_copies_wait();
    __syncthreads();

    #pragma unroll
    for (int kc = 0; kc < 2; kc++) {
      const u16* ap = &ldsA[(wave * 16 + lr) * 64 + kc * 32 + lhi * 8];
      v16bf afrag = load_frag(ap, ap + 16);
      #pragma unroll
      for (int nt = 0; nt < 4; nt++) {
        const u16* bp = &ldsBt[(nt * 16 + lr) * 64 + kc * 32 + lhi * 8];
        v16bf bfrag = load_frag(bp, bp + 16);
        acc[nt] = __builtin_amdgcn_wmma_f32_16x16x32_bf16(
            false, afrag, false, bfrag, (short)0, acc[nt], false, false);
      }
    }
  }

  // Epilogue. C layout: element j of lane -> row (lane/16)*8 + j, col lane%16.
  #pragma unroll
  for (int nt = 0; nt < 4; nt++) {
    #pragma unroll
    for (int j = 0; j < 8; j++) {
      int row = m0 + wave * 16 + lhi * 8 + j;
      int col = n0 + nt * 16 + lr;
      float v = acc[nt][j] + bias[col];
      if (EPI == EPI_QKV) {
        if (col < E_) v *= 0.125f;           // fold 1/sqrt(HD) into Q
        outB[(size_t)row * N + col] = f2bf(v);
      } else if (EPI == EPI_GELU) {
        v = 0.5f * v * (1.f + erff(v * 0.70710678118654752f));
        outB[(size_t)row * N + col] = f2bf(v);
      } else { // EPI_RES
        outF[(size_t)row * N + col] = v + res[(size_t)row * N + col];
      }
    }
  }
}

// ---------------- flash attention (causal), bf16 WMMA, f32 softmax ----------------
__global__ __launch_bounds__(128)
void attention_kernel(const u16* __restrict__ qkv, u16* __restrict__ outp)
{
  __shared__ u16 ldsK [32 * 64];      // [key][d]
  __shared__ u16 ldsVt[64 * 32];      // [d][key]
  __shared__ u16 ldsP [4][16 * 32];   // per-wave P tile [row][key]
  const int tid  = threadIdx.x, wave = tid >> 5, lane = tid & 31;
  const int lr   = lane & 15, lhi = lane >> 4;
  const int qbase = blockIdx.x * 64;
  const int bh = blockIdx.y, b = bh >> 4, h = bh & 15;
  const size_t tokbase = (size_t)b * SEQ;

  // Q fragments: 16 rows x 64 dims = 2 chunks of k=32 (direct from global; 16B aligned)
  v16bf qfrag[2];
  {
    int t = qbase + wave * 16 + lr;
    const u16* qp = qkv + (tokbase + t) * T3_ + h * HD;
    #pragma unroll
    for (int c = 0; c < 2; c++)
      qfrag[c] = load_frag(qp + c * 32 + lhi * 8, qp + c * 32 + 16 + lhi * 8);
  }

  v8f o[4];
  #pragma unroll
  for (int i = 0; i < 4; i++) o[i] = v8f_zero();
  float mrow[8], lsum[8];
  #pragma unroll
  for (int j = 0; j < 8; j++) { mrow[j] = -1e30f; lsum[j] = 0.f; }

  const int nkt = (qbase + 64) / 32;
  for (int kt = 0; kt < nkt; kt++) {
    const int kbase = kt * 32;
    __syncthreads();
    // K tile 32x64: 2x 16B async copies per thread
    #pragma unroll
    for (int q = 0; q < 2; q++) {
      int c = tid * 2 + q;                 // 256 chunks
      int key = c >> 3, off = (c & 7) * 8;
      copy16_g2l(qkv + (tokbase + kbase + key) * T3_ + E_ + h * HD + off,
                 &ldsK[key * 64 + off]);
    }
    // V tile transposed -> [d][key], pair-packed b32 stores
    #pragma unroll
    for (int i = 0; i < 8; i++) {
      int lin = tid + 128 * i;             // 0..1023
      int d = lin & 63, key = (lin >> 6) * 2;
      const u16* vp = qkv + (tokbase + kbase + key) * T3_ + 2 * E_ + h * HD + d;
      *(unsigned*)&ldsVt[d * 32 + key] =
          (unsigned)vp[0] | ((unsigned)vp[T3_] << 16);
    }
    async_copies_wait();
    __syncthreads();

    // scores: 16 queries x 32 keys, K-dim = 64 -> 2x2 WMMAs
    v8f s[2]; s[0] = v8f_zero(); s[1] = v8f_zero();
    #pragma unroll
    for (int ct = 0; ct < 2; ct++) {
      #pragma unroll
      for (int c = 0; c < 2; c++) {
        const u16* kp = &ldsK[(ct * 16 + lr) * 64 + c * 32 + lhi * 8];
        v16bf kf = load_frag(kp, kp + 16);
        s[ct] = __builtin_amdgcn_wmma_f32_16x16x32_bf16(
            false, qfrag[c], false, kf, (short)0, s[ct], false, false);
      }
    }

    // online softmax (rows live across the 16-lane halves; shfl_xor 1..8 reduces)
    #pragma unroll
    for (int j = 0; j < 8; j++) {
      const int rg = qbase + wave * 16 + lhi * 8 + j;
      float s0 = s[0][j]; if (kbase + lr > rg)      s0 = -1e30f;
      float s1 = s[1][j]; if (kbase + 16 + lr > rg) s1 = -1e30f;
      float sm = fmaxf(s0, s1);
      #pragma unroll
      for (int off = 8; off > 0; off >>= 1) sm = fmaxf(sm, __shfl_xor(sm, off, 32));
      const float mnew  = fmaxf(mrow[j], sm);
      const float scale = __expf(mrow[j] - mnew);
      const float p0 = __expf(s0 - mnew), p1 = __expf(s1 - mnew);
      float rs = p0 + p1;
      #pragma unroll
      for (int off = 8; off > 0; off >>= 1) rs += __shfl_xor(rs, off, 32);
      lsum[j] = lsum[j] * scale + rs;
      mrow[j] = mnew;
      #pragma unroll
      for (int nt = 0; nt < 4; nt++) o[nt][j] *= scale;
      ldsP[wave][(lhi * 8 + j) * 32 + lr]      = f2bf(p0);
      ldsP[wave][(lhi * 8 + j) * 32 + 16 + lr] = f2bf(p1);
    }
    __syncthreads();   // P visible; re-layout C->A through LDS

    // O += P(16x32) @ V(32x64)
    const u16* pp = &ldsP[wave][lr * 32 + lhi * 8];
    v16bf pf = load_frag(pp, pp + 16);
    #pragma unroll
    for (int nt = 0; nt < 4; nt++) {
      const u16* vp = &ldsVt[(nt * 16 + lr) * 32 + lhi * 8];
      v16bf vf = load_frag(vp, vp + 16);
      o[nt] = __builtin_amdgcn_wmma_f32_16x16x32_bf16(
          false, pf, false, vf, (short)0, o[nt], false, false);
    }
  }

  // normalize + store bf16 [token][h*64+d]
  #pragma unroll
  for (int j = 0; j < 8; j++) {
    const float inv = 1.f / lsum[j];
    const int t = qbase + wave * 16 + lhi * 8 + j;
    const size_t rowoff = (tokbase + t) * E_ + h * HD;
    #pragma unroll
    for (int nt = 0; nt < 4; nt++)
      outp[rowoff + nt * 16 + lr] = f2bf(o[nt][j] * inv);
  }
}

// ---------------- launch ----------------
extern "C" void kernel_launch(void* const* d_in, const int* in_sizes, int n_in,
                              void* d_out, int out_size, void* d_ws, size_t ws_size,
                              hipStream_t stream)
{
  (void)in_sizes; (void)n_in; (void)out_size; (void)ws_size;
  const float* x      = (const float*)d_in[0];
  const float* ln1_g  = (const float*)d_in[1];
  const float* ln1_b  = (const float*)d_in[2];
  const float* w_attn = (const float*)d_in[3];
  const float* b_attn = (const float*)d_in[4];
  const float* w_ap   = (const float*)d_in[5];
  const float* b_ap   = (const float*)d_in[6];
  const float* ln2_g  = (const float*)d_in[7];
  const float* ln2_b  = (const float*)d_in[8];
  const float* w_fc   = (const float*)d_in[9];
  const float* b_fc   = (const float*)d_in[10];
  const float* w_proj = (const float*)d_in[11];
  const float* b_proj = (const float*)d_in[12];
  float* out = (float*)d_out;

  char* ws = (char*)d_ws;
  size_t off = 0;
  auto take = [&](size_t bytes) -> char* {
    char* p = ws + off;
    off = (off + bytes + 255) & ~(size_t)255;
    return p;
  };
  u16*   xn   = (u16*)  take((size_t)NTOK * E_ * 2);       // ln1 out (bf16)
  u16*   qkv  = (u16*)  take((size_t)NTOK * T3_ * 2);      // qkv (bf16, q pre-scaled)
  u16*   attn = (u16*)  take((size_t)NTOK * E_ * 2);       // attention out (bf16)
  float* x1   = (float*)take((size_t)NTOK * E_ * 4);       // residual 1 (f32)
  u16*   hn   = (u16*)  take((size_t)NTOK * E_ * 2);       // ln2 out (bf16)
  u16*   hact = (u16*)  take((size_t)NTOK * 4 * E_ * 2);   // gelu(fc) (bf16)

  layernorm_bf16_kernel<<<NTOK, 256, 0, stream>>>(x, ln1_g, ln1_b, xn);

  gemm_bf16_kernel<EPI_QKV><<<dim3(T3_ / 64, NTOK / 64), 128, 0, stream>>>(
      xn, w_attn, b_attn, nullptr, qkv, nullptr, NTOK, T3_, E_);

  attention_kernel<<<dim3(SEQ / 64, BATCH * HEADS), 128, 0, stream>>>(qkv, attn);

  gemm_bf16_kernel<EPI_RES><<<dim3(E_ / 64, NTOK / 64), 128, 0, stream>>>(
      attn, w_ap, b_ap, x, nullptr, x1, NTOK, E_, E_);

  layernorm_bf16_kernel<<<NTOK, 256, 0, stream>>>(x1, ln2_g, ln2_b, hn);

  gemm_bf16_kernel<EPI_GELU><<<dim3(4 * E_ / 64, NTOK / 64), 128, 0, stream>>>(
      hn, w_fc, b_fc, nullptr, hact, nullptr, NTOK, 4 * E_, E_);

  gemm_bf16_kernel<EPI_RES><<<dim3(E_ / 64, NTOK / 64), 128, 0, stream>>>(
      hact, w_proj, b_proj, x1, nullptr, out, NTOK, E_, 4 * E_);
}
